// HGRNBitBlock_455266534018
// MI455X (gfx1250) — 8-bit matrix operand order repaired and checked by reading, compiled, not run on hardware
//
#include <hip/hip_runtime.h>
#include <hip/hip_bf16.h>
#include <stdint.h>


typedef __attribute__((ext_vector_type(8))) int v8i;

#define BB 4
#define LL 4096
#define HH 1024
#define II 2816
#define TT (BB * LL)
#define NC 16
#define CL (LL / NC)
#define EPSF 1e-6f

__device__ __forceinline__ float wredsum(float v) {
#pragma unroll
  for (int o = 16; o > 0; o >>= 1) v += __shfl_down(v, o, 32);
  return v;
}
__device__ __forceinline__ float wredmax(float v) {
#pragma unroll
  for (int o = 16; o > 0; o >>= 1) v = fmaxf(v, __shfl_down(v, o, 32));
  return v;
}
__device__ __forceinline__ float bredsum(float v, float* sh) {
  const int lane = threadIdx.x & 31, wid = threadIdx.x >> 5, nw = blockDim.x >> 5;
  v = wredsum(v);
  __syncthreads();
  if (lane == 0) sh[wid] = v;
  __syncthreads();
  if (wid == 0) {
    float r = (lane < nw) ? sh[lane] : 0.f;
    r = wredsum(r);
    if (lane == 0) sh[0] = r;
  }
  __syncthreads();
  return sh[0];
}
__device__ __forceinline__ float bredmax(float v, float* sh) {
  const int lane = threadIdx.x & 31, wid = threadIdx.x >> 5, nw = blockDim.x >> 5;
  v = wredmax(v);
  __syncthreads();
  if (lane == 0) sh[wid] = v;
  __syncthreads();
  if (wid == 0) {
    float r = (lane < nw) ? sh[lane] : 0.f;
    r = wredmax(r);
    if (lane == 0) sh[0] = r;
  }
  __syncthreads();
  return sh[0];
}
__device__ __forceinline__ float sigf(float x) { return 1.f / (1.f + __expf(-x)); }

__global__ __launch_bounds__(256) void k_wabs(const float* __restrict__ w, size_t n,
                                              float* __restrict__ sum) {
  __shared__ float sh[32];
  float s = 0.f;
  for (size_t i = blockIdx.x * 256ull + threadIdx.x; i < n; i += (size_t)gridDim.x * 256ull)
    s += fabsf(w[i]);
  s = bredsum(s, sh);
  if (threadIdx.x == 0) atomicAdd(sum, s);
}

__global__ __launch_bounds__(256) void k_wquant(const float* __restrict__ w, size_t n,
                                                const float* __restrict__ sum,
                                                int8_t* __restrict__ wq,
                                                float* __restrict__ wscale) {
  const float mean = fmaxf(sum[0] / (float)n, 1e-5f);
  const float ws = 1.f / mean;
  for (size_t i = blockIdx.x * 256ull + threadIdx.x; i < n; i += (size_t)gridDim.x * 256ull) {
    float q = fminf(fmaxf(rintf(w[i] * ws), -1.f), 1.f);
    wq[i] = (int8_t)q;
  }
  if (blockIdx.x == 0 && threadIdx.x == 0) wscale[0] = mean;
}

template <int E>
__global__ __launch_bounds__(256) void k_norm2_quant(const float* __restrict__ x,
                                                     const float* __restrict__ add,
                                                     const float* __restrict__ nw, int K,
                                                     int8_t* __restrict__ q,
                                                     float* __restrict__ sdeq,
                                                     float* __restrict__ resid_out) {
  __shared__ float sh[32];
  const size_t t = blockIdx.x;
  const float* xp = x + t * (size_t)K;
  float v[E];
  float ss = 0.f;
#pragma unroll
  for (int e = 0; e < E; ++e) {
    int j = e * 256 + threadIdx.x;
    float val = xp[j];
    if (add) val += add[t * (size_t)K + j];
    if (resid_out) resid_out[t * (size_t)K + j] = val;
    v[e] = val;
    ss += val * val;
  }
  float r1 = rsqrtf(bredsum(ss, sh) / (float)K + EPSF);
  ss = 0.f;
#pragma unroll
  for (int e = 0; e < E; ++e) {
    int j = e * 256 + threadIdx.x;
    float y = v[e] * r1 * (nw ? nw[j] : 1.f);
    v[e] = y;
    ss += y * y;
  }
  float r2 = rsqrtf(bredsum(ss, sh) / (float)K + EPSF);
  float mx = 0.f;
#pragma unroll
  for (int e = 0; e < E; ++e) {
    float y = v[e] * r2;
    v[e] = y;
    mx = fmaxf(mx, fabsf(y));
  }
  mx = fmaxf(bredmax(mx, sh), 1e-5f);
  const float s = 127.f / mx;
#pragma unroll
  for (int e = 0; e < E; ++e) {
    int j = e * 256 + threadIdx.x;
    float qv = fminf(fmaxf(rintf(v[e] * s), -128.f), 127.f);
    q[t * (size_t)K + j] = (int8_t)qv;
  }
  if (threadIdx.x == 0) sdeq[t] = mx / 127.f;
}

__global__ __launch_bounds__(256) void k_gated_norm_quant(const float* __restrict__ o,
                                                          const float* __restrict__ g,
                                                          const float* __restrict__ gw,
                                                          int8_t* __restrict__ q,
                                                          float* __restrict__ sdeq) {
  __shared__ float sh[32];
  const size_t t = blockIdx.x;
  const int K = HH;
  constexpr int E = HH / 256;
  float vo[E], vg[E];
  float ss = 0.f;
#pragma unroll
  for (int e = 0; e < E; ++e) {
    int j = e * 256 + threadIdx.x;
    vo[e] = o[t * (size_t)K + j];
    vg[e] = g[t * (size_t)K + j];
    ss += vo[e] * vo[e];
  }
  float r1 = rsqrtf(bredsum(ss, sh) / (float)K + EPSF);
  ss = 0.f;
#pragma unroll
  for (int e = 0; e < E; ++e) {
    int j = e * 256 + threadIdx.x;
    float y = vo[e] * r1 * gw[j] * (vg[e] * sigf(vg[e]));
    vo[e] = y;
    ss += y * y;
  }
  float r2 = rsqrtf(bredsum(ss, sh) / (float)K + EPSF);
  float mx = 0.f;
#pragma unroll
  for (int e = 0; e < E; ++e) {
    float y = vo[e] * r2;
    vo[e] = y;
    mx = fmaxf(mx, fabsf(y));
  }
  mx = fmaxf(bredmax(mx, sh), 1e-5f);
  const float s = 127.f / mx;
#pragma unroll
  for (int e = 0; e < E; ++e) {
    int j = e * 256 + threadIdx.x;
    float qv = fminf(fmaxf(rintf(vo[e] * s), -128.f), 127.f);
    q[t * (size_t)K + j] = (int8_t)qv;
  }
  if (threadIdx.x == 0) sdeq[t] = mx / 127.f;
}

__global__ __launch_bounds__(256) void k_swishmul_norm_quant(const float* __restrict__ z,
                                                             int8_t* __restrict__ q,
                                                             float* __restrict__ sdeq) {
  __shared__ float sh[32];
  const size_t t = blockIdx.x;
  constexpr int E = II / 256;
  const float* zp = z + t * (size_t)(2 * II);
  float v[E];
  float ss = 0.f;
#pragma unroll
  for (int e = 0; e < E; ++e) {
    int j = e * 256 + threadIdx.x;
    float gate = zp[j];
    float y = zp[II + j];
    float u = gate * sigf(gate) * y;
    v[e] = u;
    ss += u * u;
  }
  float r2 = rsqrtf(bredsum(ss, sh) / (float)II + EPSF);
  float mx = 0.f;
#pragma unroll
  for (int e = 0; e < E; ++e) {
    float y = v[e] * r2;
    v[e] = y;
    mx = fmaxf(mx, fabsf(y));
  }
  mx = fmaxf(bredmax(mx, sh), 1e-5f);
  const float s = 127.f / mx;
#pragma unroll
  for (int e = 0; e < E; ++e) {
    int j = e * 256 + threadIdx.x;
    float qv = fminf(fmaxf(rintf(v[e] * s), -128.f), 127.f);
    q[t * (size_t)II + j] = (int8_t)qv;
  }
  if (threadIdx.x == 0) sdeq[t] = mx / 127.f;
}

__device__ __forceinline__ v8i load_afrag(const int8_t* ap) {
  const int2 a0 = *(const int2*)(ap);
  const int2 a1 = *(const int2*)(ap + 8);
  const int2 a2 = *(const int2*)(ap + 32);
  const int2 a3 = *(const int2*)(ap + 40);
  v8i a;
  a[0] = a0.x; a[1] = a0.y; a[2] = a1.x; a[3] = a1.y;
  a[4] = a2.x; a[5] = a2.y; a[6] = a3.x; a[7] = a3.y;
  return a;
}
__device__ __forceinline__ v8i load_bfrag(const int8_t* bp) {
  const int4 b0 = *(const int4*)(bp);
  const int4 b1 = *(const int4*)(bp + 32);
  v8i b;
  b[0] = b0.x; b[1] = b0.y; b[2] = b0.z; b[3] = b0.w;
  b[4] = b1.x; b[5] = b1.y; b[6] = b1.z; b[7] = b1.w;
  return b;
}

__global__ __launch_bounds__(256) void k_bitgemm(const int8_t* __restrict__ A,
                                                 const float* __restrict__ sA,
                                                 const int8_t* __restrict__ Bw,
                                                 const float* __restrict__ sW, int K, int O,
                                                 float* __restrict__ C,
                                                 const float* __restrict__ add) {
  const int lane = threadIdx.x & 31;
  const int wave = threadIdx.x >> 5;
  const int tilesN4 = O >> 6;
  const int tile = blockIdx.x * 8 + wave;
  const int tm = tile / tilesN4;
  const int tn = tile - tm * tilesN4;
  const int row0 = tm << 4;
  const int col0 = tn << 6;

  const int8_t* ap = A + (size_t)(row0 + (lane & 15)) * K + ((lane >> 4) << 4);
  const int8_t* bp = Bw + (size_t)(col0 + (lane & 15)) * K + ((lane >> 4) << 4);
  const size_t bstride = (size_t)16 * K;

  v8i acc[4];
#pragma unroll
  for (int n = 0; n < 4; ++n) acc[n] = v8i{0, 0, 0, 0, 0, 0, 0, 0};

  const int steps = K >> 6;
  for (int s = 0; s < steps; ++s) {
    __builtin_prefetch(ap + 64, 0, 0);
    __builtin_prefetch(bp + 64, 0, 0);
    const v8i a = load_afrag(ap);
#pragma unroll
    for (int n = 0; n < 4; ++n) {
      const v8i b = load_bfrag(bp + n * bstride);
      acc[n] = __builtin_amdgcn_wmma_i32_16x16x64_iu8(true, a, true, b, acc[n], false, false);
    }
    ap += 64;
    bp += 64;
  }

  const float wsc = sW[0];
  const int rbase = row0 + ((lane >> 4) << 3);
#pragma unroll
  for (int n = 0; n < 4; ++n) {
    const int col = col0 + 16 * n + (lane & 15);
#pragma unroll
    for (int r = 0; r < 8; ++r) {
      const int row = rbase + r;
      float f = (float)acc[n][r] * sA[row] * wsc;
      const size_t o = (size_t)row * O + col;
      if (add) f += add[o];
      C[o] = f;
    }
  }
}

__global__ __launch_bounds__(256) void k_bitgemm_if(const int8_t* __restrict__ A,
                                                    const float* __restrict__ sA,
                                                    const int8_t* __restrict__ Wi,
                                                    const int8_t* __restrict__ Wf,
                                                    const float* __restrict__ sWi,
                                                    const float* __restrict__ sWf,
                                                    const float* __restrict__ lb,
                                                    float* __restrict__ Ci,
                                                    float* __restrict__ Cf) {
  const int K = HH, O = HH;
  const int lane = threadIdx.x & 31;
  const int wave = threadIdx.x >> 5;
  const int tilesN2 = O >> 5;
  const int tile = blockIdx.x * 8 + wave;
  const int tm = tile / tilesN2;
  const int tn = tile - tm * tilesN2;
  const int row0 = tm << 4;
  const int col0 = tn << 5;

  const int8_t* ap = A + (size_t)(row0 + (lane & 15)) * K + ((lane >> 4) << 4);
  const size_t boff = (size_t)(col0 + (lane & 15)) * K + ((lane >> 4) << 4);
  const int8_t* bpi = Wi + boff;
  const int8_t* bpf = Wf + boff;
  const size_t bstride = (size_t)16 * K;

  v8i acci[2], accf[2];
#pragma unroll
  for (int n = 0; n < 2; ++n) {
    acci[n] = v8i{0, 0, 0, 0, 0, 0, 0, 0};
    accf[n] = v8i{0, 0, 0, 0, 0, 0, 0, 0};
  }

  const int steps = K >> 6;
  for (int s = 0; s < steps; ++s) {
    __builtin_prefetch(ap + 64, 0, 0);
    __builtin_prefetch(bpi + 64, 0, 0);
    __builtin_prefetch(bpf + 64, 0, 0);
    const v8i a = load_afrag(ap);
#pragma unroll
    for (int n = 0; n < 2; ++n) {
      const v8i bi = load_bfrag(bpi + n * bstride);
      acci[n] = __builtin_amdgcn_wmma_i32_16x16x64_iu8(true, a, true, bi, acci[n], false, false);
      const v8i bf = load_bfrag(bpf + n * bstride);
      accf[n] = __builtin_amdgcn_wmma_i32_16x16x64_iu8(true, a, true, bf, accf[n], false, false);
    }
    ap += 64;
    bpi += 64;
    bpf += 64;
  }

  const float wi = sWi[0], wf = sWf[0];
  const int rbase = row0 + ((lane >> 4) << 3);
#pragma unroll
  for (int n = 0; n < 2; ++n) {
    const int col = col0 + 16 * n + (lane & 15);
    const float l = lb[col];
#pragma unroll
    for (int r = 0; r < 8; ++r) {
      const int row = rbase + r;
      const float sa = sA[row];
      const float iraw = (float)acci[n][r] * sa * wi;
      const float fraw = (float)accf[n][r] * sa * wf;
      const float f = l + (1.f - l) * sigf(fraw);
      const float iw = iraw * sigf(iraw) * (1.f - f);
      const size_t o = (size_t)row * O + col;
      Ci[o] = iw;
      Cf[o] = f;
    }
  }
}

__global__ __launch_bounds__(256) void k_scan1(const float* __restrict__ ib,
                                               const float* __restrict__ fb,
                                               float* __restrict__ ob,
                                               float* __restrict__ pb,
                                               float* __restrict__ chH,
                                               float* __restrict__ chF) {
  const int tid = blockIdx.x * 256 + threadIdx.x;
  const int d = tid & (HH - 1);
  const int bc = tid >> 10;
  const int c = bc & (NC - 1);
  const int b = bc >> 4;
  float h = 0.f, p = 1.f;
  for (int t = c * CL; t < (c + 1) * CL; ++t) {
    const size_t idx = ((size_t)b * LL + t) * HH + d;
    const float f = fb[idx];
    h = f * h + ib[idx];
    p *= f;
    ob[idx] = h;
    pb[idx] = p;
  }
  chH[((size_t)b * NC + c) * HH + d] = h;
  chF[((size_t)b * NC + c) * HH + d] = p;
}

__global__ __launch_bounds__(256) void k_scan2(const float* __restrict__ chH,
                                               const float* __restrict__ chF,
                                               float* __restrict__ carry) {
  const int tid = blockIdx.x * 256 + threadIdx.x;
  const int d = tid & (HH - 1);
  const int b = tid >> 10;
  float c = 0.f;
  for (int k = 0; k < NC; ++k) {
    const size_t idx = ((size_t)b * NC + k) * HH + d;
    carry[idx] = c;
    c = chH[idx] + chF[idx] * c;
  }
}

__global__ __launch_bounds__(256) void k_scan3(float* __restrict__ ob,
                                               const float* __restrict__ pb,
                                               const float* __restrict__ carry) {
  const size_t n = (size_t)TT * HH;
  for (size_t idx = blockIdx.x * 256ull + threadIdx.x; idx < n;
       idx += (size_t)gridDim.x * 256ull) {
    const int d = (int)(idx & (HH - 1));
    const size_t bt = idx >> 10;
    const int t = (int)(bt & (LL - 1));
    const int b = (int)(bt >> 12);
    const int c = t / CL;
    ob[idx] += carry[((size_t)b * NC + c) * HH + d] * pb[idx];
  }
}

extern "C" void kernel_launch(void* const* d_in, const int* in_sizes, int n_in,
                              void* d_out, int out_size, void* d_ws, size_t ws_size,
                              hipStream_t stream) {
  (void)in_sizes; (void)n_in; (void)out_size; (void)ws_size;
  const float* hidden = (const float*)d_in[0];
  const float* lb     = (const float*)d_in[1];
  const float* attn_w = (const float*)d_in[2];
  const float* w_i    = (const float*)d_in[3];
  const float* w_f    = (const float*)d_in[4];
  const float* w_g    = (const float*)d_in[5];
  const float* gn_w   = (const float*)d_in[6];
  const float* w_o    = (const float*)d_in[7];
  const float* mlp_w  = (const float*)d_in[8];
  const float* w_gate = (const float*)d_in[9];
  const float* w_down = (const float*)d_in[10];
  float* out = (float*)d_out;

  char* cur = (char*)d_ws;
  auto alloc = [&](size_t bytes) -> void* {
    void* p = (void*)cur;
    cur += (bytes + 255) & ~(size_t)255;
    return p;
  };

  int8_t* xq = (int8_t*)alloc((size_t)TT * HH);
  int8_t* oq = (int8_t*)alloc((size_t)TT * HH);
  int8_t* hq = (int8_t*)alloc((size_t)TT * HH);
  int8_t* uq = (int8_t*)alloc((size_t)TT * II);
  float* sX = (float*)alloc((size_t)TT * 4);
  float* sO = (float*)alloc((size_t)TT * 4);
  float* sH = (float*)alloc((size_t)TT * 4);
  float* sU = (float*)alloc((size_t)TT * 4);

  const float* Wsrc[6] = {w_i, w_f, w_g, w_o, w_gate, w_down};
  const size_t Wn[6] = {(size_t)HH * HH, (size_t)HH * HH, (size_t)HH * HH,
                        (size_t)HH * HH, (size_t)2 * II * HH, (size_t)HH * II};
  int8_t* Wq[6];
  for (int j = 0; j < 6; ++j) Wq[j] = (int8_t*)alloc(Wn[j]);
  float* wsum = (float*)alloc(256);
  float* wscale = (float*)alloc(256);

  float* i_buf = (float*)alloc((size_t)TT * HH * 4);
  float* f_buf = (float*)alloc((size_t)TT * HH * 4);
  float* o_buf = (float*)alloc((size_t)TT * HH * 4);
  float* p_buf = (float*)alloc((size_t)TT * HH * 4);
  float* g_buf = (float*)alloc((size_t)TT * HH * 4);
  float* ao_buf = (float*)alloc((size_t)TT * HH * 4);
  float* resid = (float*)alloc((size_t)TT * HH * 4);
  float* z_buf = (float*)alloc((size_t)TT * 2 * II * 4);
  float* chH = (float*)alloc((size_t)BB * NC * HH * 4);
  float* chF = (float*)alloc((size_t)BB * NC * HH * 4);
  float* carry = (float*)alloc((size_t)BB * NC * HH * 4);

  hipMemsetAsync(wsum, 0, 6 * sizeof(float), stream);
  for (int j = 0; j < 6; ++j)
    k_wabs<<<2048, 256, 0, stream>>>(Wsrc[j], Wn[j], wsum + j);
  for (int j = 0; j < 6; ++j)
    k_wquant<<<2048, 256, 0, stream>>>(Wsrc[j], Wn[j], wsum + j, Wq[j], wscale + j);

  k_norm2_quant<HH / 256><<<TT, 256, 0, stream>>>(hidden, nullptr, attn_w, HH, xq, sX, nullptr);

  const int tilesM = TT / 16;

  k_bitgemm_if<<<tilesM * (HH / 32) / 8, 256, 0, stream>>>(
      xq, sX, Wq[0], Wq[1], wscale + 0, wscale + 1, lb, i_buf, f_buf);

  k_bitgemm<<<tilesM * (HH / 64) / 8, 256, 0, stream>>>(xq, sX, Wq[2], wscale + 2, HH, HH, g_buf, nullptr);

  k_scan1<<<(BB * HH * NC) / 256, 256, 0, stream>>>(i_buf, f_buf, o_buf, p_buf, chH, chF);
  k_scan2<<<(BB * HH) / 256, 256, 0, stream>>>(chH, chF, carry);
  k_scan3<<<4096, 256, 0, stream>>>(o_buf, p_buf, carry);

  k_gated_norm_quant<<<TT, 256, 0, stream>>>(o_buf, g_buf, gn_w, oq, sO);
  k_bitgemm<<<tilesM * (HH / 64) / 8, 256, 0, stream>>>(oq, sO, Wq[3], wscale + 3, HH, HH, ao_buf, nullptr);

  k_norm2_quant<HH / 256><<<TT, 256, 0, stream>>>(ao_buf, hidden, mlp_w, HH, hq, sH, resid);

  k_bitgemm<<<tilesM * (2 * II / 64) / 8, 256, 0, stream>>>(hq, sH, Wq[4], wscale + 4, HH, 2 * II, z_buf, nullptr);
  k_swishmul_norm_quant<<<TT, 256, 0, stream>>>(z_buf, uq, sU);
  k_bitgemm<<<tilesM * (HH / 64) / 8, 256, 0, stream>>>(uq, sU, Wq[5], wscale + 5, II, HH, out, resid);
}
